// NormDepthConv2d_63582695850752
// MI455X (gfx1250) — matrix operand order repaired and checked by reading, not compiled, not run on hardware
//
#include <hip/hip_runtime.h>

typedef __attribute__((ext_vector_type(16))) _Float16 v16h;
typedef __attribute__((ext_vector_type(8)))  float    v8f;

#define KSZ    5
#define PADR   2
#define HIMG   480
#define WIMG   640
#define OC     64
#define KK     25                  // 5*5, padded to 32 for WMMA K
#define TILE_H 8
#define TILE_W 128
#define LSTR   (TILE_W + 2*PADR)   // 132 floats per staged row
#define LROWS  (TILE_H + 2*PADR)   // 12 staged rows
#define STAGE_N (LROWS * LSTR)     // 1584 staged elements

typedef __attribute__((address_space(1))) int gbl_i32;
typedef __attribute__((address_space(3))) int lds_i32;

// SWAPX16: group-of-32 swizzle, xor_mask=0x10, and_mask=0x1f
__device__ __forceinline__ float swapx16(float v) {
    int r = __builtin_amdgcn_ds_swizzle(__builtin_bit_cast(int, v), 0x401F);
    return __builtin_bit_cast(float, r);
}

// out[b,o,y,x] = sum_k W[o,k] * ((xref - patch_k) * mask_k)
// 64x25 weight matmul per pixel via V_WMMA_F32_16X16X32_F16 (K padded to 32).
__global__ __launch_bounds__(256)
void norm_depth_conv_wmma(const float* __restrict__ x,
                          const float* __restrict__ w,
                          float* __restrict__ out)
{
    __shared__ float tile[STAGE_N];

    const int b    = blockIdx.z;
    const int y0   = blockIdx.y * TILE_H;
    const int x0   = blockIdx.x * TILE_W;
    const int tid  = threadIdx.x;
    const int wave = tid >> 5;        // 0..7 : row within tile
    const int lane = tid & 31;
    const int lx   = lane & 15;       // N index within a 16-pixel group
    const int khalf = lane >> 4;      // which K-half this lane holds in A/B/D

    const size_t HW = (size_t)HIMG * WIMG;
    const float* xb = x + (size_t)b * HW;

    __builtin_prefetch(w, 0, 1);      // tiny, heavily reused weights

    // ---- stage zero-padded input tile into LDS ----
#if __has_builtin(__builtin_amdgcn_global_load_async_to_lds_b32)
    // Async path: in-bounds elements DMA'd global->LDS (ASYNCcnt); OOB pad
    // elements written as zeros through the (disjoint-address) DS path.
    for (int i = tid; i < STAGE_N; i += 256) {
        int ty = i / LSTR;
        int tx = i - ty * LSTR;
        int gy = y0 + ty - PADR;
        int gx = x0 + tx - PADR;
        if ((unsigned)gy < (unsigned)HIMG && (unsigned)gx < (unsigned)WIMG) {
            __builtin_amdgcn_global_load_async_to_lds_b32(
                (gbl_i32*)(xb + (size_t)gy * WIMG + gx),
                (lds_i32*)&tile[i],
                0, 0);
        } else {
            tile[i] = 0.0f;
        }
    }
    asm volatile("s_wait_asynccnt 0x0" ::: "memory");
#else
    for (int i = tid; i < STAGE_N; i += 256) {
        int ty = i / LSTR;
        int tx = i - ty * LSTR;
        int gy = y0 + ty - PADR;
        int gx = x0 + tx - PADR;
        float v = 0.0f;
        if ((unsigned)gy < (unsigned)HIMG && (unsigned)gx < (unsigned)WIMG)
            v = xb[(size_t)gy * WIMG + gx];
        tile[i] = v;
    }
#endif

    // ---- A fragments (weights -> f16, K padded 25->32 with zeros)
    // A k order used here (contiguous, same as B): lanes 0-15 hold K {0..15}, lanes 16-31
    // hold K {16..31}; element e -> K = e + 16*khalf.
    v16h afrag[4];
    #pragma unroll
    for (int t = 0; t < 4; ++t) {
        const int m = t * 16 + lx;
        #pragma unroll
        for (int e = 0; e < 16; ++e) {
            int k0 = e;
            int k  = khalf ? (k0 + 16) : k0;
            float wv = (k < KK) ? w[m * KK + k] : 0.0f;
            afrag[t][e] = (_Float16)wv;
        }
    }

    // per-tile uniform store bases, biased +4 channels so (r-4)*HW*4 bytes
    // fits the signed 24-bit instruction offset (-4.9MB..+3.7MB).
    float* tbase[4];
    #pragma unroll
    for (int t = 0; t < 4; ++t)
        tbase[t] = out + ((size_t)b * OC + 16 * t + 4) * HW;

    __syncthreads();

    const int y = y0 + wave;

    // each iteration: one pixel per lane -> 32 pixels, 2 WMMA groups
    for (int cg = 0; cg < TILE_W / 32; ++cg) {
        const int px = cg * 32 + lane;     // this lane's pixel column in tile

        // ---- gather own 5x5 patch + masked mean (once per pixel) ----
        float patch[KK];
        float sum = 0.0f, cnt = 0.0f;
        #pragma unroll
        for (int k = 0; k < KK; ++k) {
            const int ky = k / KSZ, kx = k % KSZ;
            float v = tile[(wave + ky) * LSTR + (px + kx)];
            patch[k] = v;
            sum += v;
            cnt += (v > 1e-4f) ? 1.0f : 0.0f;
        }
        const float xref = sum / (cnt + 1e-6f);

        float nb[32];
        #pragma unroll
        for (int k = 0; k < KK; ++k)
            nb[k] = (patch[k] > 1e-4f) ? (xref - patch[k]) : 0.0f;
        #pragma unroll
        for (int k = KK; k < 32; ++k) nb[k] = 0.0f;

        // ---- cross-half exchange: low lanes send nb[16+j], high lanes send
        // nb[j]; after SWAPX16 each lane holds the partner slice it needs.
        float sx[16];
        #pragma unroll
        for (int j = 0; j < 16; ++j)
            sx[j] = swapx16(khalf ? nb[j] : nb[16 + j]);

        // ---- B fragments (32x16 f16): lane L holds N=L&15; lanes 0-15 carry
        // K=0..15, lanes 16-31 carry K=16..31 (element e -> K = e + 16*khalf).
        // Group A = pixels cg*32+0..15, group B = pixels cg*32+16..31.
        v16h bfragA, bfragB;
        #pragma unroll
        for (int e = 0; e < 16; ++e) {
            bfragA[e] = (_Float16)(khalf ? sx[e]       : nb[e]);
            bfragB[e] = (_Float16)(khalf ? nb[16 + e]  : sx[e]);
        }

        // ---- 8 WMMAs -> 64 channels x 32 pixels, streamed out NT ----
        const size_t voff = (size_t)y * WIMG + (size_t)(x0 + cg * 32 + lx)
                          + (size_t)khalf * (8 * HW);
        #pragma unroll
        for (int t = 0; t < 4; ++t) {
            v8f cA = {}, cB = {};
            cA = __builtin_amdgcn_wmma_f32_16x16x32_f16(
                     false, afrag[t], false, bfragA, (short)0, cA, false, false);
            cB = __builtin_amdgcn_wmma_f32_16x16x32_f16(
                     false, afrag[t], false, bfragB, (short)0, cB, false, false);
            float* pA = tbase[t] + voff;    // channel 16t+4+8*khalf, pixel col
            float* pB = pA + 16;
            #pragma unroll
            for (int r = 0; r < 8; ++r) {
                const ptrdiff_t choff = (ptrdiff_t)(r - 4) * (ptrdiff_t)HW;
                __builtin_nontemporal_store(cA[r], pA + choff);
                __builtin_nontemporal_store(cB[r], pB + choff);
            }
        }
    }
}

extern "C" void kernel_launch(void* const* d_in, const int* in_sizes, int n_in,
                              void* d_out, int out_size, void* d_ws, size_t ws_size,
                              hipStream_t stream)
{
    const float* x = (const float*)d_in[0];   // [B,1,480,640] f32
    const float* w = (const float*)d_in[1];   // [1,64,5,5]    f32
    float* out = (float*)d_out;               // [B,64,480,640] f32

    const int B = in_sizes[0] / (HIMG * WIMG);
    dim3 grid(WIMG / TILE_W, HIMG / TILE_H, B);
    norm_depth_conv_wmma<<<grid, dim3(256), 0, stream>>>(x, w, out);
}
